// MultiHeadSlotAttention_4355096838595
// MI455X (gfx1250) — matrix operand order repaired and checked by reading, not compiled, not run on hardware
//
#include <hip/hip_runtime.h>
#include <hip/hip_bf16.h>

// ---------------------------------------------------------------------------
// MultiHeadSlotAttention for MI455X (gfx1250, wave32, WMMA + TDM)
// B=32 N=4096 D=256 H=4 K=8 ITERS=3 DH=64 HID=512
// ---------------------------------------------------------------------------

#define Bsz   32
#define Nsz   4096
#define Dsz   256
#define Hsz   4
#define Ksl   8
#define DHsz  64
#define HIDsz 512
#define ITERS 3
#define SCALE 0.125f           // DH^-0.5
#define EPSd  1e-8f
#define LNEPS 1e-5f
#define Mrows (Bsz*Nsz)        // 131072

typedef __attribute__((ext_vector_type(16))) __bf16 v16bf;
typedef __attribute__((ext_vector_type(8)))  __bf16 v8bf;
typedef __attribute__((ext_vector_type(8)))  float  v8f;

typedef unsigned int u32;
typedef __attribute__((ext_vector_type(4))) u32 u32x4;
typedef __attribute__((ext_vector_type(8))) int i32x8;
typedef __attribute__((ext_vector_type(4))) int i32x4;

union BF16x16 { v16bf v; v8bf h[2]; __bf16 e[16]; };
union BF16x8  { v8bf  v; __bf16 e[8]; };
union F32x8   { v8f   v; float  f[8]; };

#define WMMA_BF16(Af, Bf, Cf) \
  __builtin_amdgcn_wmma_f32_16x16x32_bf16(false, (Af), false, (Bf), (short)0, (Cf), false, false)

#if defined(__has_builtin)
#if __has_builtin(__builtin_amdgcn_tensor_load_to_lds)
#define HAVE_TDM 1
#endif
#endif

// ---------------------------------------------------------------------------
// slots = mu + exp(logsigma) * noise
// ---------------------------------------------------------------------------
__global__ void k_init(const float* __restrict__ mu, const float* __restrict__ ls,
                       const float* __restrict__ noise, float* __restrict__ slots, int n) {
  int i = blockIdx.x * 256 + threadIdx.x;
  if (i < n) {
    int d = i & (Dsz - 1);
    slots[i] = mu[d] + __expf(ls[d]) * noise[i];
  }
}

__global__ void k_zero(float* __restrict__ p, int n) {
  int i = blockIdx.x * blockDim.x + threadIdx.x;
  if (i < n) p[i] = 0.f;
}

// W [256,256] fp32 -> WT [n][k] bf16 (transposed, for contiguous B-fragment loads)
__global__ void k_wt(const float* __restrict__ W, __bf16* __restrict__ WT) {
  int i = blockIdx.x * 256 + threadIdx.x;        // 65536 total
  int r = i >> 8, c = i & 255;
  WT[(size_t)c * 256 + r] = (__bf16)W[(size_t)r * 256 + c];
}

// ---------------------------------------------------------------------------
// LayerNorm(inputs) -> xhat bf16 [M,256].  One wave per row, 8 elems per lane.
// ---------------------------------------------------------------------------
__global__ __launch_bounds__(256) void k_ln(const float* __restrict__ x,
                                            const float* __restrict__ g,
                                            const float* __restrict__ bb,
                                            __bf16* __restrict__ xhat) {
  int row  = blockIdx.x * 8 + (threadIdx.x >> 5);
  int lane = threadIdx.x & 31;
  const float* xr = x + (size_t)row * Dsz + lane * 8;
  float v[8], s = 0.f, s2 = 0.f;
#pragma unroll
  for (int i = 0; i < 8; ++i) { v[i] = xr[i]; s += v[i]; }
#pragma unroll
  for (int m = 1; m < 32; m <<= 1) s += __shfl_xor(s, m);
  float mean = s * (1.f / Dsz);
#pragma unroll
  for (int i = 0; i < 8; ++i) { float d = v[i] - mean; s2 += d * d; }
#pragma unroll
  for (int m = 1; m < 32; m <<= 1) s2 += __shfl_xor(s2, m);
  float inv = rsqrtf(s2 * (1.f / Dsz) + LNEPS);
  BF16x8 o;
#pragma unroll
  for (int i = 0; i < 8; ++i) {
    int c = lane * 8 + i;
    o.e[i] = (__bf16)((v[i] - mean) * inv * g[c] + bb[c]);
  }
  *(v8bf*)(xhat + (size_t)row * Dsz + lane * 8) = o.v;
}

// ---------------------------------------------------------------------------
// Fused projection GEMM: xhat[M,256] @ {Wk, Wv} in one pass.
//  - TDM stages the block's 32x256 bf16 A-tile into LDS once (8x less global
//    A traffic than per-wave loads across both outputs).
//  - 8 waves: (rowTile 0/1) x (colGroup 0..3); each wave owns 16 rows x 64
//    cols of BOTH outputs: 8 WMMA accumulators (64 VGPRs).
//  - K-features stored row-major [M,256]; V-features stored transposed
//    [B*H*64, N] so the attention kernel's B-fragments are contiguous.
// ---------------------------------------------------------------------------
__global__ __launch_bounds__(256) void k_proj(const __bf16* __restrict__ X,
                                              const __bf16* __restrict__ WkT,
                                              const __bf16* __restrict__ WvT,
                                              __bf16* __restrict__ kOut,
                                              __bf16* __restrict__ vT) {
  __shared__ alignas(16) __bf16 sX[32 * Dsz];    // 16 KiB tile
  int lane = threadIdx.x & 31;
  int w    = threadIdx.x >> 5;
  int rowBase0 = blockIdx.x * 32;

#ifdef HAVE_TDM
  if (w == 0) {
    // Tensor DMA descriptor (D#), 2D tile: 256 x 32 rows of bf16.
    unsigned long long ga = (unsigned long long)(size_t)(X + (size_t)rowBase0 * Dsz);
    u32 lds = (u32)(size_t)&sX[0];
    u32x4 g0;
    g0[0] = 1u;                                        // count=1, user mode
    g0[1] = lds;                                       // lds_addr
    g0[2] = (u32)ga;                                   // global_addr[31:0]
    g0[3] = (u32)((ga >> 32) & 0x01FFFFFFu) | (2u << 30); // addr[56:32] | type=2
    i32x8 g1;
    g1[0] = (int)(1u << 16);                           // data_size=1 (2 bytes)
    g1[1] = (int)(256u << 16);                         // tensor_dim0 = 256 (lo16)
    g1[2] = 0;                                         // dim0 hi, dim1 lo
    g1[3] = (int)(2u | (256u << 16));                  // dim1 hi (131072), tile_dim0=256
    g1[4] = 32;                                        // tile_dim1=32, tile_dim2=0
    g1[5] = 256;                                       // tensor_dim0_stride = 256
    g1[6] = 0;
    g1[7] = 0;
    i32x4 gz4 = {0, 0, 0, 0};
    i32x8 gz8 = {0, 0, 0, 0, 0, 0, 0, 0};
    __builtin_amdgcn_tensor_load_to_lds(g0, g1, gz4, gz4, gz8, 0);
    __builtin_amdgcn_s_wait_tensorcnt(0);
  }
#else
  // fallback: cooperative vector copy (64 B per thread)
  {
    const v8bf* src = (const v8bf*)(X + (size_t)rowBase0 * Dsz);
    v8bf* dst = (v8bf*)&sX[0];
#pragma unroll
    for (int i = 0; i < 4; ++i)
      dst[threadIdx.x + i * 256] = src[threadIdx.x + i * 256];
  }
#endif
  __syncthreads();

  int rowTile  = w & 1;
  int colGroup = w >> 1;
  int l15   = lane & 15;
  int koffA = (lane < 16) ? 0 : 16;
  int koffB = (lane < 16) ? 0 : 16;
  const __bf16* xrow = sX + (size_t)(rowTile * 16 + l15) * Dsz;

  v8f accK[4] = { {0,0,0,0,0,0,0,0}, {0,0,0,0,0,0,0,0},
                  {0,0,0,0,0,0,0,0}, {0,0,0,0,0,0,0,0} };
  v8f accV[4] = { {0,0,0,0,0,0,0,0}, {0,0,0,0,0,0,0,0},
                  {0,0,0,0,0,0,0,0}, {0,0,0,0,0,0,0,0} };

  for (int kb = 0; kb < Dsz; kb += 32) {
    BF16x16 A;
    A.h[0] = *(const v8bf*)(xrow + kb + koffA);        // ds_load_b128
    A.h[1] = *(const v8bf*)(xrow + kb + 8 + koffA);
#pragma unroll
    for (int s = 0; s < 4; ++s) {
      int n = colGroup * 64 + s * 16 + l15;
      const __bf16* wpk = WkT + (size_t)n * Dsz + kb + koffB;
      const __bf16* wpv = WvT + (size_t)n * Dsz + kb + koffB;
      __builtin_prefetch(wpk + 64, 0, 1);              // global_prefetch_b8
      BF16x16 Bk, Bv;
      Bk.h[0] = *(const v8bf*)(wpk);
      Bk.h[1] = *(const v8bf*)(wpk + 8);
      Bv.h[0] = *(const v8bf*)(wpv);
      Bv.h[1] = *(const v8bf*)(wpv + 8);
      accK[s] = WMMA_BF16(A.v, Bk.v, accK[s]);
      accV[s] = WMMA_BF16(A.v, Bv.v, accV[s]);
    }
  }

  int rowBase = rowBase0 + rowTile * 16;
  // K-features row-major [token][256] bf16
#pragma unroll
  for (int s = 0; s < 4; ++s) {
    F32x8 t; t.v = accK[s];
    int col = colGroup * 64 + s * 16 + l15;
#pragma unroll
    for (int r = 0; r < 8; ++r) {
      int m = r + ((lane < 16) ? 0 : 8);
      kOut[(size_t)(rowBase + m) * Dsz + col] = (__bf16)t.f[r];
    }
  }
  // V-features transposed [(b*H + h)*64 + dh][N]; 8 consecutive n per lane
  {
    int bI = rowBase >> 12;                       // / N
    int n0 = (rowBase & (Nsz - 1)) + ((lane < 16) ? 0 : 8);
#pragma unroll
    for (int s = 0; s < 4; ++s) {
      F32x8 t; t.v = accV[s];
      int col = colGroup * 64 + s * 16 + l15;
      int h = col >> 6, dh = col & 63;
      BF16x8 o;
#pragma unroll
      for (int r = 0; r < 8; ++r) o.e[r] = (__bf16)t.f[r];
      *(v8bf*)(vT + ((size_t)((bI * Hsz + h) * DHsz + dh)) * Nsz + n0) = o.v;
    }
  }
}

// ---------------------------------------------------------------------------
// q = LN(slots) @ Wq, packed for WMMA A-fragments: qpad[(b*H+h)*16 + k][64] bf16
// (rows 8..15 of each 16x64 tile are padding; A-matrix padding rows only
//  affect unused output rows, so they need no zeroing)
// ---------------------------------------------------------------------------
__global__ __launch_bounds__(256) void k_qproj(const float* __restrict__ slots,
                                               const float* __restrict__ g_sl,
                                               const float* __restrict__ b_sl,
                                               const float* __restrict__ Wq,
                                               __bf16* __restrict__ qpad) {
  int row = blockIdx.x;               // b*8 + k
  int b = row >> 3, k = row & 7;
  int tid = threadIdx.x;
  __shared__ float sX[Dsz], red[Dsz];
  float x = slots[(size_t)row * Dsz + tid];
  red[tid] = x; __syncthreads();
  for (int off = 128; off; off >>= 1) { if (tid < off) red[tid] += red[tid + off]; __syncthreads(); }
  float mean = red[0] * (1.f / Dsz); __syncthreads();
  float dev = x - mean;
  red[tid] = dev * dev; __syncthreads();
  for (int off = 128; off; off >>= 1) { if (tid < off) red[tid] += red[tid + off]; __syncthreads(); }
  float var = red[0] * (1.f / Dsz);
  sX[tid] = dev * rsqrtf(var + LNEPS) * g_sl[tid] + b_sl[tid];
  __syncthreads();
  float s = 0.f;
  for (int d = 0; d < Dsz; ++d) s += sX[d] * Wq[(size_t)d * Dsz + tid];
  int h = tid >> 6, dh = tid & 63;
  qpad[((size_t)(b * Hsz + h) * 16 + k) * DHsz + dh] = (__bf16)s;
}

// ---------------------------------------------------------------------------
// Attention: one wave per block; SPLIT=32 blocks per (b,h), each handles 128 n.
// dots = Q(16x64) . K^T via 2 WMMAs per 16-col tile; softmax over 8 slots is
// lane-local; attn repacked to an A-fragment through LDS, then 4 WMMAs
// accumulate upd[16x64].  Partials combined with fp32 global atomics.
// ---------------------------------------------------------------------------
#define SPLIT 32
#define NCHUNK (Nsz / SPLIT / 32)   // 4 chunks of 32 columns

__global__ __launch_bounds__(32) void k_attn(const __bf16* __restrict__ qpad,
                                             const __bf16* __restrict__ kfeat,
                                             const __bf16* __restrict__ vT,
                                             float* __restrict__ updp,
                                             float* __restrict__ denom) {
  int lane = threadIdx.x;
  int bh = blockIdx.x / SPLIT;
  int sp = blockIdx.x % SPLIT;
  int b = bh >> 2, h = bh & 3;
  int l15   = lane & 15;
  int koffA = (lane < 16) ? 0 : 16;
  int koffB = (lane < 16) ? 0 : 16;

  __shared__ alignas(16) __bf16 sA[16 * 32];

  const __bf16* qb = qpad + (size_t)bh * 16 * DHsz;
  BF16x16 aQ[2];
#pragma unroll
  for (int t = 0; t < 2; ++t) {
    const __bf16* qp = qb + l15 * DHsz + t * 32 + koffA;
    aQ[t].h[0] = *(const v8bf*)(qp);
    aQ[t].h[1] = *(const v8bf*)(qp + 8);
  }

  v8f accU[4] = { {0,0,0,0,0,0,0,0}, {0,0,0,0,0,0,0,0},
                  {0,0,0,0,0,0,0,0}, {0,0,0,0,0,0,0,0} };
  float dsum[8] = {0.f,0.f,0.f,0.f,0.f,0.f,0.f,0.f};

  int nStart = sp * (Nsz / SPLIT);
  for (int ch = 0; ch < NCHUNK; ++ch) {
    int nc = nStart + ch * 32;
#pragma unroll
    for (int t = 0; t < 2; ++t) {
      int n = nc + t * 16 + l15;
      const __bf16* kp = kfeat + ((size_t)(b * Nsz + n)) * Dsz + h * DHsz;
      BF16x16 bK0, bK1;
      bK0.h[0] = *(const v8bf*)(kp + koffB);
      bK0.h[1] = *(const v8bf*)(kp + koffB + 8);
      bK1.h[0] = *(const v8bf*)(kp + 32 + koffB);
      bK1.h[1] = *(const v8bf*)(kp + 32 + koffB + 8);
      v8f c = {0,0,0,0,0,0,0,0};
      c = WMMA_BF16(aQ[0].v, bK0.v, c);
      c = WMMA_BF16(aQ[1].v, bK1.v, c);
      // competitive softmax over the 8 slots (valid data in lanes 0..15)
      F32x8 tt; tt.v = c;
      float mx = -3.0e38f;
#pragma unroll
      for (int r = 0; r < 8; ++r) { tt.f[r] *= SCALE; mx = fmaxf(mx, tt.f[r]); }
      float ex[8], se = 0.f;
#pragma unroll
      for (int r = 0; r < 8; ++r) { ex[r] = __expf(tt.f[r] - mx); se += ex[r]; }
      float inv = 1.f / se;
      if (lane < 16) {
#pragma unroll
        for (int r = 0; r < 8; ++r) {
          float a = ex[r] * inv;
          dsum[r] += a;
          sA[r * 32 + t * 16 + l15]       = (__bf16)a;
          sA[(r + 8) * 32 + t * 16 + l15] = (__bf16)0.f;  // zero pad rows
        }
      }
    }
    __syncthreads();
    BF16x16 aA;
    const __bf16* ap = sA + l15 * 32 + koffA;
    aA.h[0] = *(const v8bf*)(ap);
    aA.h[1] = *(const v8bf*)(ap + 8);
#pragma unroll
    for (int s = 0; s < 4; ++s) {
      int d = s * 16 + l15;
      const __bf16* vp = vT + ((size_t)bh * DHsz + d) * Nsz + nc + koffB;
      BF16x16 bV;
      bV.h[0] = *(const v8bf*)(vp);
      bV.h[1] = *(const v8bf*)(vp + 8);
      accU[s] = WMMA_BF16(aA.v, bV.v, accU[s]);
    }
    __syncthreads();
  }

#pragma unroll
  for (int k = 0; k < 8; ++k) {
    float v = dsum[k];
    v += __shfl_xor(v, 1);
    v += __shfl_xor(v, 2);
    v += __shfl_xor(v, 4);
    v += __shfl_xor(v, 8);
    if (lane == 0) atomicAdd(&denom[bh * Ksl + k], v);
  }
  if (lane < 16) {
#pragma unroll
    for (int s = 0; s < 4; ++s) {
      F32x8 t; t.v = accU[s];
#pragma unroll
      for (int r = 0; r < 8; ++r)
        atomicAdd(&updp[((size_t)bh * 16 + r) * DHsz + s * 16 + l15], t.f[r]);
    }
  }
}

// ---------------------------------------------------------------------------
// Finalize one slot row: gather heads, /denom, @Wc, GRU cell, LN, MLP residual.
// All fp32 (tiny: 256 rows). One block (256 threads) per (b,k).
// ---------------------------------------------------------------------------
__global__ __launch_bounds__(256) void k_final(
    const float* __restrict__ slots_in, float* __restrict__ slots_out,
    const float* __restrict__ updp, const float* __restrict__ denom,
    const float* __restrict__ Wc,
    const float* __restrict__ wih, const float* __restrict__ whh,
    const float* __restrict__ bih, const float* __restrict__ bhh,
    const float* __restrict__ W1, const float* __restrict__ b1,
    const float* __restrict__ W2, const float* __restrict__ b2,
    const float* __restrict__ g_ml, const float* __restrict__ b_ml) {
  int row = blockIdx.x;             // b*8 + k
  int b = row >> 3, k = row & 7;
  int tid = threadIdx.x;
  __shared__ float sU[Dsz], sH[Dsz], sUc[Dsz], sX[Dsz], sHid[HIDsz], red[Dsz];

  int hD = tid >> 6, dh = tid & 63;
  float den = denom[(b * Hsz + hD) * Ksl + k] + EPSd;
  sU[tid] = updp[((size_t)(b * Hsz + hD) * 16 + k) * DHsz + dh] / den;
  sH[tid] = slots_in[(size_t)row * Dsz + tid];
  __syncthreads();

  {
    float s = 0.f;
    for (int d = 0; d < Dsz; ++d) s += sU[d] * Wc[(size_t)d * Dsz + tid];
    sUc[tid] = s;
  }
  __syncthreads();

  float gi_r = bih[tid],         gh_r = bhh[tid];
  float gi_z = bih[Dsz + tid],   gh_z = bhh[Dsz + tid];
  float gi_n = bih[2*Dsz + tid], gh_n = bhh[2*Dsz + tid];
  for (int d = 0; d < Dsz; ++d) {
    float uc = sUc[d], hh = sH[d];
    gi_r += uc * wih[(size_t)tid * Dsz + d];
    gh_r += hh * whh[(size_t)tid * Dsz + d];
    gi_z += uc * wih[(size_t)(Dsz + tid) * Dsz + d];
    gh_z += hh * whh[(size_t)(Dsz + tid) * Dsz + d];
    gi_n += uc * wih[(size_t)(2*Dsz + tid) * Dsz + d];
    gh_n += hh * whh[(size_t)(2*Dsz + tid) * Dsz + d];
  }
  float r  = 1.f / (1.f + __expf(-(gi_r + gh_r)));
  float z  = 1.f / (1.f + __expf(-(gi_z + gh_z)));
  float nc = tanhf(gi_n + r * gh_n);
  float snew = (1.f - z) * nc + z * sH[tid];

  red[tid] = snew; __syncthreads();
  for (int off = 128; off; off >>= 1) { if (tid < off) red[tid] += red[tid + off]; __syncthreads(); }
  float mean = red[0] * (1.f / Dsz); __syncthreads();
  float dev = snew - mean;
  red[tid] = dev * dev; __syncthreads();
  for (int off = 128; off; off >>= 1) { if (tid < off) red[tid] += red[tid + off]; __syncthreads(); }
  float var = red[0] * (1.f / Dsz);
  sX[tid] = dev * rsqrtf(var + LNEPS) * g_ml[tid] + b_ml[tid];
  __syncthreads();

  for (int jj = 0; jj < 2; ++jj) {
    int j = tid + jj * Dsz;
    float s = b1[j];
    for (int d = 0; d < Dsz; ++d) s += sX[d] * W1[(size_t)d * HIDsz + j];
    sHid[j] = fmaxf(s, 0.f);
  }
  __syncthreads();

  float o = snew + b2[tid];
  for (int j = 0; j < HIDsz; ++j) o += sHid[j] * W2[(size_t)j * Dsz + tid];
  slots_out[(size_t)row * Dsz + tid] = o;
}

// ---------------------------------------------------------------------------
// Host side
// ---------------------------------------------------------------------------
extern "C" void kernel_launch(void* const* d_in, const int* in_sizes, int n_in,
                              void* d_out, int out_size, void* d_ws, size_t ws_size,
                              hipStream_t stream) {
  const float* inputs   = (const float*)d_in[0];
  const float* noise    = (const float*)d_in[1];
  const float* slots_mu = (const float*)d_in[2];
  const float* slots_ls = (const float*)d_in[3];
  const float* g_in = (const float*)d_in[4];
  const float* b_in = (const float*)d_in[5];
  const float* g_sl = (const float*)d_in[6];
  const float* b_sl = (const float*)d_in[7];
  const float* g_ml = (const float*)d_in[8];
  const float* b_ml = (const float*)d_in[9];
  const float* Wq = (const float*)d_in[10];
  const float* Wk = (const float*)d_in[11];
  const float* Wv = (const float*)d_in[12];
  const float* Wc = (const float*)d_in[13];
  const float* wih = (const float*)d_in[14];
  const float* whh = (const float*)d_in[15];
  const float* bih = (const float*)d_in[16];
  const float* bhh = (const float*)d_in[17];
  const float* W1 = (const float*)d_in[18];
  const float* b1 = (const float*)d_in[19];
  const float* W2 = (const float*)d_in[20];
  const float* b2 = (const float*)d_in[21];

  // workspace layout
  char* w = (char*)d_ws;
  const size_t FEAT = (size_t)Mrows * Dsz * sizeof(__bf16);   // 64 MiB
  __bf16* xhat  = (__bf16*)(w);
  __bf16* kfeat = (__bf16*)(w + FEAT);
  __bf16* vT    = (__bf16*)(w + 2 * FEAT);
  __bf16* wkT   = (__bf16*)(w + 3 * FEAT);
  __bf16* wvT   = wkT + Dsz * Dsz;
  __bf16* qpad  = wvT + Dsz * Dsz;                            // B*H*16*64
  float*  updp  = (float*)((char*)(qpad + Bsz * Hsz * 16 * DHsz));
  float*  denom = updp + Bsz * Hsz * 16 * DHsz;               // B*H*8 (+pad)
  float*  slots = denom + 1024;

  k_init<<<(Bsz*Ksl*Dsz + 255)/256, 256, 0, stream>>>(slots_mu, slots_ls, noise, slots, Bsz*Ksl*Dsz);
  k_ln<<<Mrows/8, 256, 0, stream>>>(inputs, g_in, b_in, xhat);
  k_wt<<<Dsz*Dsz/256, 256, 0, stream>>>(Wk, wkT);
  k_wt<<<Dsz*Dsz/256, 256, 0, stream>>>(Wv, wvT);
  k_proj<<<Mrows/32, 256, 0, stream>>>(xhat, wkT, wvT, kfeat, vT);

  for (int it = 0; it < ITERS; ++it) {
    k_zero<<<(Bsz*Hsz*16*DHsz + 255)/256, 256, 0, stream>>>(updp, Bsz*Hsz*16*DHsz);
    k_zero<<<4, 256, 0, stream>>>(denom, 1024);
    k_qproj<<<Bsz*Ksl, 256, 0, stream>>>(slots, g_sl, b_sl, Wq, qpad);
    k_attn<<<Bsz*Hsz*SPLIT, 32, 0, stream>>>(qpad, kfeat, vT, updp, denom);
    float* sout = (it == ITERS - 1) ? (float*)d_out : slots;
    k_final<<<Bsz*Ksl, 256, 0, stream>>>(slots, sout, updp, denom, Wc,
                                         wih, whh, bih, bhh, W1, b1, W2, b2, g_ml, b_ml);
  }
}
